// HybridQLSTM_65481071407182
// MI455X (gfx1250) — matrix operand order repaired and checked by reading, not compiled, not run on hardware
//
#include <hip/hip_runtime.h>
#include <cstddef>
#include <cstdint>

#define GAMMA 1.0f
#define T_STEPS 256
#define BATCH   128
#define DIN     1024
#define HID     1024
#define KDIM    2048            // D + H
#define NT      (HID / 16)      // 64 n-tiles
#define KT      (KDIM / 32)     // 64 k-tiles of 32
#define FRAG_ELEMS 512          // 32 lanes * 16 bf16 per B/A fragment
#define FRAG_BYTES 1024
#define SLAB_BYTES (4 * FRAG_BYTES)                 // 4 gates per k-slab
#define GATE_STRIDE ((size_t)NT * KT * FRAG_ELEMS)  // packed elems per gate

typedef __attribute__((ext_vector_type(16))) __bf16 v16bf;
typedef __attribute__((ext_vector_type(8)))  __bf16 v8bf;
typedef __attribute__((ext_vector_type(8)))  float  v8f;

__device__ __forceinline__ float sigmoidf_(float x) {
    return 1.0f / (1.0f + __expf(-x));
}

// One wave moves 512B (32 lanes x 16B) global -> LDS through the async
// engine. Tracked by ASYNCcnt.
__device__ __forceinline__ void async_copy_512(unsigned lds_byte, const void* gptr) {
    asm volatile("global_load_async_to_lds_b128 %0, %1, off"
                 :
                 : "v"(lds_byte), "v"((unsigned long long)(uintptr_t)gptr)
                 : "memory");
}

__device__ __forceinline__ void wait_async_zero() {
    asm volatile("s_wait_asynccnt 0x0" ::: "memory");
}

// ---------------------------------------------------------------------------
// Pack fp32 W[H, 2048] (4 gates) into bf16 WMMA-B fragment order.
// Fragment (nt, kt): lane L holds column n = nt*16 + (L&15),
// K = kt*32 + ((j>>3)*16) + ((L>>4)*8) + (j&7), j = 0..15 (matches A table order).
// ---------------------------------------------------------------------------
__global__ void pack_weights(const float* __restrict__ Wf, const float* __restrict__ Wi,
                             const float* __restrict__ Wu, const float* __restrict__ Wo,
                             __bf16* __restrict__ Wpack) {
    unsigned idx  = blockIdx.x * 256u + threadIdx.x;   // 0 .. 4*NT*KT*512-1
    unsigned j    = idx & 15u;
    unsigned lane = (idx >> 4) & 31u;
    unsigned kt   = (idx >> 9) & 63u;
    unsigned nt   = (idx >> 15) & 63u;
    unsigned gate = idx >> 21;
    unsigned n    = nt * 16u + (lane & 15u);
    unsigned k    = kt * 32u + ((j >> 3) << 4) + ((lane >> 4) << 3) + (j & 7u);
    const float* W = (gate == 0) ? Wf : (gate == 1) ? Wi : (gate == 2) ? Wu : Wo;
    Wpack[idx] = (__bf16)W[(size_t)n * KDIM + k];
}

__global__ void zero_hc(float* __restrict__ hc) {
    hc[blockIdx.x * 256u + threadIdx.x] = 0.0f;
}

// ---------------------------------------------------------------------------
// Per-step: RBF term k[b] = exp(-gamma*||x-h||^2) and combined = [bf16(x), bf16(h)]
// ---------------------------------------------------------------------------
__global__ void prestep(const float* __restrict__ x_t, const float* __restrict__ hbuf,
                        __bf16* __restrict__ cmb, float* __restrict__ kbuf) {
    const int b   = blockIdx.x;
    const int tid = threadIdx.x;
    __shared__ float red[256];
    const float* xr = x_t  + (size_t)b * DIN;
    const float* hr = hbuf + (size_t)b * HID;
    __bf16*      cr = cmb  + (size_t)b * KDIM;
    float s = 0.0f;
    for (int d = tid; d < DIN; d += 256) {
        float xv = xr[d];
        float hv = hr[d];
        float df = xv - hv;
        s += df * df;
        cr[d]       = (__bf16)xv;
        cr[DIN + d] = (__bf16)hv;
    }
    red[tid] = s;
    __syncthreads();
    for (int o = 128; o > 0; o >>= 1) {
        if (tid < o) red[tid] += red[tid + o];
        __syncthreads();
    }
    if (tid == 0) kbuf[b] = __expf(-GAMMA * red[0]);
}

// ---------------------------------------------------------------------------
// Fused 4-gate step GEMM + LSTM update with async-to-LDS double buffering.
// Grid: 64 blocks (n-tiles), 256 threads = 8 waves (m-tiles covering B=128).
// Staging is divergence-free: the 4KB k-slab (4 gate fragments) is split
// across all 8 waves — wave w copies half-fragment (gate w>>1, half w&1)
// with one async b128 op (32 lanes x 16B = 512B) while everyone computes the
// other buffer via ds_load + WMMA.
// ---------------------------------------------------------------------------
__global__ void __launch_bounds__(256)
lstm_step(const __bf16* __restrict__ Wpack, const __bf16* __restrict__ cmb,
          const float* __restrict__ bf_, const float* __restrict__ bi_,
          const float* __restrict__ bu_, const float* __restrict__ bo_,
          const float* __restrict__ kbuf,
          float* __restrict__ hbuf, float* __restrict__ cbuf,
          float* __restrict__ out_t) {
    const int ntile = blockIdx.x;            // 0..63
    const int wave  = threadIdx.x >> 5;      // 0..7 == m-tile
    const int lane  = threadIdx.x & 31;
    const int mtile = wave;

    // 2 stages x (4 gates x 1KB) slabs
    __shared__ __align__(16) __bf16 smem[2][4][FRAG_ELEMS];

    // A fragment addressing (ISA 16-bit A 16x32): lane holds row m = lane&15,
    // K = kg + {0..7} and kg + {16..23}, kg = (lane>>4)*8.
    const int arow = mtile * 16 + (lane & 15);
    const int kg   = (lane >> 4) * 8;
    const __bf16* abase = cmb + (size_t)arow * KDIM + kg;

    // Staging addresses: wave w owns gate w>>1, half w&1.
    const unsigned lane16    = (unsigned)lane * 16u;
    const unsigned slab_off  = (unsigned)wave * 512u + lane16;      // into 4KB slab
    const unsigned smem_base = (unsigned)(uintptr_t)(&smem[0][0][0]);
    const char* gsrc = (const char*)(Wpack + ((size_t)(wave >> 1) * NT + ntile)
                                                 * (size_t)KT * FRAG_ELEMS)
                       + (size_t)(wave & 1) * 512u + lane16;

    // Prologue: fill buffer 0 with the kt=0 slab.
    async_copy_512(smem_base + slab_off, gsrc);
    wait_async_zero();
    __syncthreads();

    v8f accF = {}, accI = {}, accU = {}, accO = {};

    for (int kt = 0; kt < KT; ++kt) {
        const int buf = kt & 1;

        // Prefetch next k-slab through the async engine (all waves, no exec
        // masking in the hot loop).
        if (kt + 1 < KT) {
            async_copy_512(smem_base + (unsigned)(buf ^ 1) * SLAB_BYTES + slab_off,
                           gsrc + (size_t)(kt + 1) * FRAG_BYTES);
        }

        // A fragment from global (per-wave unique).
        const __bf16* ap = abase + kt * 32;
        v8bf alo = *(const v8bf*)(ap);
        v8bf ahi = *(const v8bf*)(ap + 16);
        v16bf a = __builtin_shufflevector(alo, ahi,
                      0, 1, 2, 3, 4, 5, 6, 7, 8, 9, 10, 11, 12, 13, 14, 15);

        // B fragments from LDS (lane reads its contiguous 32B).
        v16bf bF = *(const v16bf*)(&smem[buf][0][lane * 16]);
        v16bf bI = *(const v16bf*)(&smem[buf][1][lane * 16]);
        v16bf bU = *(const v16bf*)(&smem[buf][2][lane * 16]);
        v16bf bO = *(const v16bf*)(&smem[buf][3][lane * 16]);

        accF = __builtin_amdgcn_wmma_f32_16x16x32_bf16(false, a, false, bF, (short)0, accF, false, false);
        accI = __builtin_amdgcn_wmma_f32_16x16x32_bf16(false, a, false, bI, (short)0, accI, false, false);
        accU = __builtin_amdgcn_wmma_f32_16x16x32_bf16(false, a, false, bU, (short)0, accU, false, false);
        accO = __builtin_amdgcn_wmma_f32_16x16x32_bf16(false, a, false, bO, (short)0, accO, false, false);

        // Drain this wave's ASYNCcnt, then block-wide sync before flipping
        // buffers (write->read and read->write hazards both covered).
        wait_async_zero();
        __syncthreads();
    }

    // Epilogue: C/D f32 16x16 layout — VGPR r holds M=r (lanes 0-15) / M=r+8
    // (lanes 16-31), N = lane&15 in both halves.
    const int n    = ntile * 16 + (lane & 15);
    const int mofs = (lane >> 4) * 8;
    const float biasF = bf_[n];
    const float biasI = bi_[n];
    const float biasU = bu_[n];
    const float biasO = bo_[n];

#pragma unroll
    for (int r = 0; r < 8; ++r) {
        const int b = mtile * 16 + mofs + r;
        const float kk = kbuf[b];
        const float f = sigmoidf_(accF[r] + biasF + kk);
        const float i = sigmoidf_(accI[r] + biasI + kk);
        const float g = tanhf(accU[r] + biasU);
        const float o = sigmoidf_(accO[r] + biasO + kk);
        const size_t off = (size_t)b * HID + n;
        float c = cbuf[off];
        c = f * c + i * g;
        const float h = o * tanhf(c);
        cbuf[off]  = c;
        hbuf[off]  = h;
        out_t[off] = h;
    }
}

// ---------------------------------------------------------------------------
// Append hT and cT after the [T,B,H] outputs.
// ---------------------------------------------------------------------------
__global__ void finalize(const float* __restrict__ hbuf, const float* __restrict__ cbuf,
                         float* __restrict__ out_tail) {
    const int i = blockIdx.x * 256 + threadIdx.x;   // 0 .. B*H-1
    out_tail[i]               = hbuf[i];
    out_tail[BATCH * HID + i] = cbuf[i];
}

extern "C" void kernel_launch(void* const* d_in, const int* in_sizes, int n_in,
                              void* d_out, int out_size, void* d_ws, size_t ws_size,
                              hipStream_t stream) {
    (void)in_sizes; (void)n_in; (void)out_size; (void)ws_size;
    const float* inputs = (const float*)d_in[0];
    const float* Wf = (const float*)d_in[1];  const float* bf_ = (const float*)d_in[2];
    const float* Wi = (const float*)d_in[3];  const float* bi_ = (const float*)d_in[4];
    const float* Wu = (const float*)d_in[5];  const float* bu_ = (const float*)d_in[6];
    const float* Wo = (const float*)d_in[7];  const float* bo_ = (const float*)d_in[8];
    float* out = (float*)d_out;

    // Workspace layout
    char* ws = (char*)d_ws;
    __bf16* Wpack = (__bf16*)ws;                                  // 16 MB
    size_t off = 4 * GATE_STRIDE * sizeof(__bf16);
    __bf16* cmb = (__bf16*)(ws + off);                            // 512 KB
    off += (size_t)BATCH * KDIM * sizeof(__bf16);
    float* hbuf = (float*)(ws + off);                             // B*H floats
    float* cbuf = hbuf + (size_t)BATCH * HID;                     // B*H floats
    float* kbuf = cbuf + (size_t)BATCH * HID;                     // B floats

    zero_hc<<<(2 * BATCH * HID) / 256, 256, 0, stream>>>(hbuf);
    pack_weights<<<(4u * NT * KT * FRAG_ELEMS) / 256, 256, 0, stream>>>(Wf, Wi, Wu, Wo, Wpack);

    for (int t = 0; t < T_STEPS; ++t) {
        prestep<<<BATCH, 256, 0, stream>>>(inputs + (size_t)t * BATCH * DIN, hbuf, cmb, kbuf);
        lstm_step<<<NT, 256, 0, stream>>>(Wpack, cmb, bf_, bi_, bu_, bo_, kbuf,
                                          hbuf, cbuf, out + (size_t)t * BATCH * HID);
    }

    finalize<<<(BATCH * HID) / 256, 256, 0, stream>>>(hbuf, cbuf,
                                                      out + (size_t)T_STEPS * BATCH * HID);
}
